// GPT_14912126451799
// MI455X (gfx1250) — matrix operand order repaired and checked by reading, not compiled, not run on hardware
//
#include <hip/hip_runtime.h>

// ---------------------------------------------------------------------------
// GPT-2-small-ish forward (L=8, E=1024, H=16, B=2, T=1024, V=50257) on gfx1250.
// All GEMMs via v_wmma_f32_16x16x32_f16 (f16 in, fp32 accum). Weights are
// converted once to f16 *transposed* in workspace so GEMM staging is a linear
// 16B-chunk copy; per-layer f16 weights (~25MB) and the f16 head (~103MB) are
// L2-resident (192MB L2). GEMM staging uses the CDNA5 async LDS engine
// (global_load_async_to_lds_b128 / ASYNCcnt) with double-buffered LDS so DMA
// overlaps WMMA. Attention = flash-v1 in VALU fp32 (~3% of FLOPs).
// ---------------------------------------------------------------------------

typedef __attribute__((ext_vector_type(16))) _Float16 v16h;
typedef __attribute__((ext_vector_type(8)))  _Float16 v8h;
typedef __attribute__((ext_vector_type(8)))  float    v8f;

#define TM 128
#define TN 128
#define TK 32
#define HD 64
#define SLAB_BYTES (TM * TK * 2)          // 8 KB per operand slab
#define BUF_BYTES  (2 * SLAB_BYTES)       // 16 KB per double-buffer stage
#define GEMM_LDS   (2 * BUF_BYTES)        // 32 KB dynamic LDS

// ------------------------------ weight convert ------------------------------
// src: fp32 [R][C] row-major  ->  dst: f16 [C][R] row-major (i.e. transposed)
__global__ void conv_t_kernel(const float* __restrict__ src,
                              _Float16* __restrict__ dst, int R, int C) {
  size_t i = (size_t)blockIdx.x * 256 + threadIdx.x;
  size_t tot = (size_t)R * C;
  if (i >= tot) return;
  int r = (int)(i / C);
  int c = (int)(i % C);
  // weights are read exactly once here: non-temporal load keeps L2 for f16.
  float v = __builtin_nontemporal_load(src + i);
  dst[(size_t)c * R + r] = (_Float16)v;
}

// --------------------------------- embedding --------------------------------
__global__ void embed_kernel(const int* __restrict__ ctx,
                             const float* __restrict__ tok,
                             const float* __restrict__ pos,
                             float* __restrict__ x, int T, int E, size_t tot) {
  size_t i = (size_t)blockIdx.x * 256 + threadIdx.x;
  if (i >= tot) return;
  int e = (int)(i % E);
  size_t bt = i / E;
  int t = (int)(bt % T);
  int tk = ctx[bt];
  x[i] = tok[(size_t)tk * E + e] + pos[(size_t)t * E + e];
}

// --------------------------------- layernorm --------------------------------
// One block per row; writes f16 (next GEMM's A operand). b may be nullptr.
__global__ __launch_bounds__(256)
void ln_kernel(const float* __restrict__ x, const float* __restrict__ w,
               const float* __restrict__ b, _Float16* __restrict__ out, int E) {
  __shared__ float red[256];
  int row = blockIdx.x, tid = threadIdx.x;
  const float* xr = x + (size_t)row * E;
  float s = 0.f, s2 = 0.f;
  for (int i = tid; i < E; i += 256) { float v = xr[i]; s += v; s2 += v * v; }
  red[tid] = s; __syncthreads();
  for (int st = 128; st > 0; st >>= 1) { if (tid < st) red[tid] += red[tid + st]; __syncthreads(); }
  float mean = red[0] / E; __syncthreads();
  red[tid] = s2; __syncthreads();
  for (int st = 128; st > 0; st >>= 1) { if (tid < st) red[tid] += red[tid + st]; __syncthreads(); }
  float var = red[0] / E - mean * mean;
  float rs = rsqrtf(var + 1e-5f);
  _Float16* o = out + (size_t)row * E;
  for (int i = tid; i < E; i += 256) {
    float v = (xr[i] - mean) * rs * w[i];
    if (b) v += b[i];
    o[i] = (_Float16)v;
  }
}

// -------------------------- async staging helpers ---------------------------
// 16B global -> LDS copy through the async DMA engine (ASYNCcnt-tracked).
__device__ __forceinline__ void async_copy16(unsigned lds_byte_addr,
                                             const _Float16* g) {
  asm volatile("global_load_async_to_lds_b128 %0, %1, off"
               :: "v"(lds_byte_addr), "v"((unsigned long long)(size_t)g)
               : "memory");
}
__device__ __forceinline__ void wait_async_le4() {
  asm volatile("s_wait_asynccnt 0x4" ::: "memory");
}
__device__ __forceinline__ void wait_async_0() {
  asm volatile("s_wait_asynccnt 0x0" ::: "memory");
}

// ------------------------------- WMMA GEMM ----------------------------------
// C[M,N] = A[M,K](f16,row-major) x B[K,N], B given as Bt[N,K](f16,row-major).
// MODE 0: Cf = v(+bias)   1: Cf += v(+bias)   2: Ch = relu(v+bias) (f16)
// Block 256 thr = 8 waves (2x4); wave tile 64x32 = 4x2 WMMA accumulators.
// Full tiles: double-buffered async-LDS pipeline. Edge tiles (head GEMM's
// last N block): guarded synchronous staging with zero-fill.
template <int MODE>
__global__ __launch_bounds__(256)
void gemm_wmma(const _Float16* __restrict__ A, const _Float16* __restrict__ Bt,
               float* __restrict__ Cf, _Float16* __restrict__ Ch,
               const float* __restrict__ bias, int M, int N, int K) {
  extern __shared__ char smem_raw[];     // GEMM_LDS bytes
  const int tid  = threadIdx.x;
  const int lane = tid & 31;
  const int wave = tid >> 5;
  const int wrow = wave >> 2;            // 0..1 -> +64 in M
  const int wcol = wave & 3;             // 0..3 -> +32 in N
  const int m0 = blockIdx.y * TM;
  const int n0 = blockIdx.x * TN;
  const int half = lane >> 4;            // ISA 16-bit operand half-wave split
  const int l15  = lane & 15;
  const unsigned lds0 = __builtin_amdgcn_groupstaticsize();  // dyn-LDS base

  v8f acc[4][2];
#pragma unroll
  for (int i = 0; i < 4; ++i)
#pragma unroll
    for (int j = 0; j < 2; ++j) acc[i][j] = {};

  // this thread's staging chunk coordinates (2 chunks of A + 2 of B per slab)
  const int c0 = tid, c1 = tid + 256;    // chunk ids in [0,512)
  const int r0 = c0 >> 2, o0 = (c0 & 3) * 8;   // row, f16 offset within row
  const int r1 = c1 >> 2, o1 = (c1 & 3) * 8;

  const bool full = (m0 + TM <= M) && (n0 + TN <= N);
  const int nk = K / TK;

  if (full) {
    // -------- async double-buffered pipeline --------
    auto issue = [&](int kk, int buf) {
      unsigned base = lds0 + (unsigned)buf * BUF_BYTES;
      async_copy16(base + (unsigned)(r0 * TK + o0) * 2,
                   A + (size_t)(m0 + r0) * K + kk + o0);
      async_copy16(base + (unsigned)(r1 * TK + o1) * 2,
                   A + (size_t)(m0 + r1) * K + kk + o1);
      async_copy16(base + SLAB_BYTES + (unsigned)(r0 * TK + o0) * 2,
                   Bt + (size_t)(n0 + r0) * K + kk + o0);
      async_copy16(base + SLAB_BYTES + (unsigned)(r1 * TK + o1) * 2,
                   Bt + (size_t)(n0 + r1) * K + kk + o1);
    };
    issue(0, 0);
    for (int i = 0; i < nk; ++i) {
      int cur = i & 1;
      if (i + 1 < nk) { issue((i + 1) * TK, cur ^ 1); wait_async_le4(); }
      else            { wait_async_0(); }
      __syncthreads();   // slab i visible to all waves; DMA fills slab i+1

      const _Float16* sA = (const _Float16*)(smem_raw + cur * BUF_BYTES);
      const _Float16* sB = sA + TM * TK;
      v16h afrag[4], bfrag[2];
#pragma unroll
      for (int mt = 0; mt < 4; ++mt) {
        const _Float16* p = &sA[(wrow * 64 + mt * 16 + l15) * TK + half * 8];
        union { v16h v; struct { v8h lo, hi; } s; } u;
        u.s.lo = *(const v8h*)p;            // K = half*8 .. +7
        u.s.hi = *(const v8h*)(p + 16);     // K = 16+half*8 .. +7
        afrag[mt] = u.v;
      }
#pragma unroll
      for (int nt = 0; nt < 2; ++nt) {
        const _Float16* p = &sB[(wcol * 32 + nt * 16 + l15) * TK + half * 8];
        union { v16h v; struct { v8h lo, hi; } s; } u;
        u.s.lo = *(const v8h*)p;            // K = half*8 .. +7
        u.s.hi = *(const v8h*)(p + 16);     // K = 16+half*8 .. +7
        bfrag[nt] = u.v;
      }
#pragma unroll
      for (int mt = 0; mt < 4; ++mt)
#pragma unroll
        for (int nt = 0; nt < 2; ++nt)
          acc[mt][nt] = __builtin_amdgcn_wmma_f32_16x16x32_f16(
              false, afrag[mt], false, bfrag[nt], (short)0, acc[mt][nt],
              false, false);
      __syncthreads();   // all reads of buf done (dscnt flushed at barrier)
    }
  } else {
    // -------- guarded synchronous staging (edge tiles) --------
    _Float16* sA = (_Float16*)smem_raw;
    _Float16* sB = sA + TM * TK;
    for (int kk = 0; kk < K; kk += TK) {
#pragma unroll
      for (int it = 0; it < 2; ++it) {
        int r = it ? r1 : r0, o = it ? o1 : o0;
        int gm = m0 + r;
        int4 av = {0, 0, 0, 0};
        if (gm < M) av = *(const int4*)(A + (size_t)gm * K + kk + o);
        *(int4*)(&sA[r * TK + o]) = av;
        int gn = n0 + r;
        int4 bv = {0, 0, 0, 0};
        if (gn < N) bv = *(const int4*)(Bt + (size_t)gn * K + kk + o);
        *(int4*)(&sB[r * TK + o]) = bv;
        if (kk + TK < K && gn < N)
          __builtin_prefetch(Bt + (size_t)gn * K + kk + TK + o, 0, 1);
      }
      __syncthreads();
      v16h afrag[4], bfrag[2];
#pragma unroll
      for (int mt = 0; mt < 4; ++mt) {
        const _Float16* p = &sA[(wrow * 64 + mt * 16 + l15) * TK + half * 8];
        union { v16h v; struct { v8h lo, hi; } s; } u;
        u.s.lo = *(const v8h*)p;
        u.s.hi = *(const v8h*)(p + 16);
        afrag[mt] = u.v;
      }
#pragma unroll
      for (int nt = 0; nt < 2; ++nt) {
        const _Float16* p = &sB[(wcol * 32 + nt * 16 + l15) * TK + half * 8];
        union { v16h v; struct { v8h lo, hi; } s; } u;
        u.s.lo = *(const v8h*)p;
        u.s.hi = *(const v8h*)(p + 16);
        bfrag[nt] = u.v;
      }
#pragma unroll
      for (int mt = 0; mt < 4; ++mt)
#pragma unroll
        for (int nt = 0; nt < 2; ++nt)
          acc[mt][nt] = __builtin_amdgcn_wmma_f32_16x16x32_f16(
              false, afrag[mt], false, bfrag[nt], (short)0, acc[mt][nt],
              false, false);
      __syncthreads();
    }
  }

  // Epilogue: C/D layout = lane holds column N=l15 (+half selects M rows+8)
#pragma unroll
  for (int mt = 0; mt < 4; ++mt) {
#pragma unroll
    for (int nt = 0; nt < 2; ++nt) {
      int gn = n0 + wcol * 32 + nt * 16 + l15;
      if (gn >= N) continue;
      float bv = bias ? bias[gn] : 0.f;
#pragma unroll
      for (int r = 0; r < 8; ++r) {
        int gm = m0 + wrow * 64 + mt * 16 + r + half * 8;
        if (gm >= M) continue;
        float v = acc[mt][nt][r] + bv;
        if (MODE == 0)      Cf[(size_t)gm * N + gn] = v;
        else if (MODE == 1) Cf[(size_t)gm * N + gn] += v;
        else                Ch[(size_t)gm * N + gn] = (_Float16)fmaxf(v, 0.f);
      }
    }
  }
}

// ------------------------------ flash attention -----------------------------
// qkv: fp32 [B*T, 3E]; one query per thread; 64-key K/V tiles in LDS;
// online softmax; writes y as f16 [B*T, E] (feeds proj GEMM).
__global__ __launch_bounds__(128)
void attn_kernel(const float* __restrict__ qkv, _Float16* __restrict__ y16,
                 int B, int T, int H, int E) {
  const int QB = 128;
  int blk  = blockIdx.x;
  int qblk = blk % (T / QB);
  int bh   = blk / (T / QB);
  int h = bh % H, b = bh / H;
  int tid = threadIdx.x;
  int qi  = qblk * QB + tid;
  const int E3 = 3 * E;
  const float scale = 0.125f;  // 1/sqrt(64)

  __shared__ float sK[64][HD];
  __shared__ float sV[64][HD];

  float q[HD], acc[HD];
  const float* qrow = qkv + (size_t)(b * T + qi) * E3 + h * HD;
#pragma unroll
  for (int d = 0; d < HD; ++d) { q[d] = qrow[d] * scale; acc[d] = 0.f; }
  float m = -1e30f, l = 0.f;

  int ntiles = (qblk * QB + QB - 1) / 64 + 1;  // block-uniform causal bound
  for (int t0 = 0; t0 < ntiles; ++t0) {
    __syncthreads();
    for (int c = tid; c < 64 * (HD / 4); c += QB) {
      int j  = c / (HD / 4);
      int d4 = (c % (HD / 4)) * 4;
      size_t base = (size_t)(b * T + t0 * 64 + j) * E3 + h * HD + d4;
      *(float4*)&sK[j][d4] = *(const float4*)&qkv[base + E];
      *(float4*)&sV[j][d4] = *(const float4*)&qkv[base + 2 * E];
    }
    __syncthreads();
    int jmax = qi - t0 * 64;
    if (jmax > 63) jmax = 63;
    for (int j = 0; j <= jmax; ++j) {
      float s = 0.f;
#pragma unroll
      for (int d = 0; d < HD; ++d) s += q[d] * sK[j][d];
      float mn = fmaxf(m, s);
      float corr = __expf(m - mn);
      float p = __expf(s - mn);
      l = l * corr + p;
#pragma unroll
      for (int d = 0; d < HD; ++d) acc[d] = acc[d] * corr + p * sV[j][d];
      m = mn;
    }
  }
  float inv = 1.f / l;
  _Float16* yrow = y16 + (size_t)(b * T + qi) * E + h * HD;
#pragma unroll
  for (int d = 0; d < HD; ++d) yrow[d] = (_Float16)(acc[d] * inv);
}

// ---------------------------------- loss ------------------------------------
__global__ void zero_kernel(float* p) { *p = 0.f; }

__global__ __launch_bounds__(256)
void loss_kernel(const float* __restrict__ logits, const int* __restrict__ tgt,
                 float* __restrict__ loss, int V, float invBT) {
  __shared__ float red[256];
  int row = blockIdx.x, tid = threadIdx.x;
  const float* lr = logits + (size_t)row * V;
  float mx = -1e30f;
  for (int i = tid; i < V; i += 256) mx = fmaxf(mx, lr[i]);
  red[tid] = mx; __syncthreads();
  for (int st = 128; st > 0; st >>= 1) { if (tid < st) red[tid] = fmaxf(red[tid], red[tid + st]); __syncthreads(); }
  mx = red[0]; __syncthreads();
  float s = 0.f;
  for (int i = tid; i < V; i += 256) s += __expf(lr[i] - mx);
  red[tid] = s; __syncthreads();
  for (int st = 128; st > 0; st >>= 1) { if (tid < st) red[tid] += red[tid + st]; __syncthreads(); }
  if (tid == 0) {
    float lse = mx + logf(red[0]);
    atomicAdd(loss, (lse - lr[tgt[row]]) * invBT);
  }
}

// --------------------------------- launcher ---------------------------------
extern "C" void kernel_launch(void* const* d_in, const int* in_sizes, int n_in,
                              void* d_out, int out_size, void* d_ws, size_t ws_size,
                              hipStream_t stream) {
  (void)in_sizes; (void)n_in; (void)out_size; (void)ws_size;
  const int Lc = 8, Ec = 1024, Hc = 16, Vc = 50257, Bc = 2, Tc = 1024;
  const int BT = Bc * Tc;

  const int*   ctx    = (const int*)d_in[0];
  const int*   tgt    = (const int*)d_in[1];
  const float* tok    = (const float*)d_in[2];
  const float* pos    = (const float*)d_in[3];
  const float* w_qkv  = (const float*)d_in[4];
  const float* w_proj = (const float*)d_in[5];
  const float* ln1_w  = (const float*)d_in[6];
  const float* ln2_w  = (const float*)d_in[7];
  const float* w_fc   = (const float*)d_in[8];
  const float* b_fc   = (const float*)d_in[9];
  const float* w_fc2  = (const float*)d_in[10];
  const float* b_fc2  = (const float*)d_in[11];
  const float* lnf_w  = (const float*)d_in[12];
  const float* lnf_b  = (const float*)d_in[13];
  const float* head_w = (const float*)d_in[14];
  const float* head_b = (const float*)d_in[15];
  float* out = (float*)d_out;

  // workspace carve-out
  char* cur = (char*)d_ws;
  auto alloc = [&](size_t bytes) {
    void* p = cur; cur += (bytes + 255) & ~(size_t)255; return p;
  };
  _Float16* wqkvT  = (_Float16*)alloc((size_t)Lc * 3 * Ec * Ec * 2);
  _Float16* wprojT = (_Float16*)alloc((size_t)Lc * Ec * Ec * 2);
  _Float16* wfcT   = (_Float16*)alloc((size_t)Lc * 4 * Ec * Ec * 2);
  _Float16* wfc2T  = (_Float16*)alloc((size_t)Lc * Ec * 4 * Ec * 2);
  _Float16* headT  = (_Float16*)alloc((size_t)Vc * Ec * 2);
  float*    x      = (float*)   alloc((size_t)BT * Ec * 4);
  _Float16* h16    = (_Float16*)alloc((size_t)BT * Ec * 2);
  float*    qkv    = (float*)   alloc((size_t)BT * 3 * Ec * 4);
  _Float16* y16    = (_Float16*)alloc((size_t)BT * Ec * 2);
  _Float16* fc16   = (_Float16*)alloc((size_t)BT * 4 * Ec * 2);

  auto launchT = [&](const float* src, _Float16* dst, int R, int C) {
    size_t tot = (size_t)R * C;
    conv_t_kernel<<<(unsigned)((tot + 255) / 256), 256, 0, stream>>>(src, dst, R, C);
  };
  auto gemm = [&](int mode, const _Float16* A, const _Float16* Bt, float* Cf,
                  _Float16* Ch, const float* bias, int M, int N, int K) {
    dim3 g((N + TN - 1) / TN, (M + TM - 1) / TM), blk(256);
    if (mode == 0)      gemm_wmma<0><<<g, blk, GEMM_LDS, stream>>>(A, Bt, Cf, Ch, bias, M, N, K);
    else if (mode == 1) gemm_wmma<1><<<g, blk, GEMM_LDS, stream>>>(A, Bt, Cf, Ch, bias, M, N, K);
    else                gemm_wmma<2><<<g, blk, GEMM_LDS, stream>>>(A, Bt, Cf, Ch, bias, M, N, K);
  };

  // ---- one-time f16 transposed weights (recomputed each call: stateless) ----
  launchT(head_w, headT, Ec, Vc);
  for (int l = 0; l < Lc; ++l) {
    launchT(w_qkv  + (size_t)l * Ec * 3 * Ec, wqkvT  + (size_t)l * 3 * Ec * Ec, Ec, 3 * Ec);
    launchT(w_proj + (size_t)l * Ec * Ec,     wprojT + (size_t)l * Ec * Ec,     Ec, Ec);
    launchT(w_fc   + (size_t)l * Ec * 4 * Ec, wfcT   + (size_t)l * 4 * Ec * Ec, Ec, 4 * Ec);
    launchT(w_fc2  + (size_t)l * 4 * Ec * Ec, wfc2T  + (size_t)l * Ec * 4 * Ec, 4 * Ec, Ec);
  }

  // ---- embedding ----
  {
    size_t tot = (size_t)BT * Ec;
    embed_kernel<<<(unsigned)((tot + 255) / 256), 256, 0, stream>>>(
        ctx, tok, pos, x, Tc, Ec, tot);
  }

  // ---- transformer blocks ----
  for (int l = 0; l < Lc; ++l) {
    ln_kernel<<<BT, 256, 0, stream>>>(x, ln1_w + (size_t)l * Ec, nullptr, h16, Ec);
    gemm(0, h16, wqkvT + (size_t)l * 3 * Ec * Ec, qkv, nullptr, nullptr,
         BT, 3 * Ec, Ec);
    attn_kernel<<<Bc * Hc * (Tc / 128), 128, 0, stream>>>(qkv, y16, Bc, Tc, Hc, Ec);
    gemm(1, y16, wprojT + (size_t)l * Ec * Ec, x, nullptr, nullptr, BT, Ec, Ec);
    ln_kernel<<<BT, 256, 0, stream>>>(x, ln2_w + (size_t)l * Ec, nullptr, h16, Ec);
    gemm(2, h16, wfcT + (size_t)l * 4 * Ec * Ec, nullptr, fc16,
         b_fc + (size_t)l * 4 * Ec, BT, 4 * Ec, Ec);
    gemm(1, fc16, wfc2T + (size_t)l * Ec * 4 * Ec, x, nullptr,
         b_fc2 + (size_t)l * Ec, BT, Ec, 4 * Ec);
  }

  // ---- final LN + head -> logits in d_out ----
  ln_kernel<<<BT, 256, 0, stream>>>(x, lnf_w, lnf_b, h16, Ec);
  gemm(0, h16, headT, out, nullptr, head_b, BT, Vc, Ec);

  // ---- mean NLL into d_out[BT*V] ----
  float* loss = out + (size_t)BT * Vc;
  zero_kernel<<<1, 1, 0, stream>>>(loss);
  loss_kernel<<<BT, 256, 0, stream>>>(out, tgt, loss, Vc, 1.f / BT);
}
